// MultiheadSelfAttention_75110388073013
// MI455X (gfx1250) — compile-verified, matrix operand order checked by reading, not run on hardware
//
#include <hip/hip_runtime.h>
#include <hip/hip_bf16.h>

typedef __bf16 bf16;
typedef __attribute__((ext_vector_type(16))) __bf16 bf16x16;
typedef __attribute__((ext_vector_type(8)))  float  f32x8;

#define LDA 40   // GEMM staging pitch (bf16): 80B rows -> 16B-aligned chunks, 16-bank spread
#define LDK 72   // K/V tile pitch: 144B rows -> 16B-aligned chunks, 16-bank spread
#define LDP 36   // P bounce tile pitch

// ---- CDNA5 async (memory -> LDS, ASYNCcnt-tracked) with safe fallback ------
#if defined(__AMDGCN__) && __has_builtin(__builtin_amdgcn_global_load_async_to_lds_b128)
#define USE_ASYNC_LDS 1
#define AS1 __attribute__((address_space(1)))
#define AS3 __attribute__((address_space(3)))
typedef int v4i __attribute__((vector_size(16)));   // builtin's pointee type
#define ASYNC_LD16(g, l) \
  __builtin_amdgcn_global_load_async_to_lds_b128((AS1 v4i*)(g), (AS3 v4i*)(l), 0, 0)
#if __has_builtin(__builtin_amdgcn_s_wait_asynccnt)
#define ASYNC_WAIT(n) __builtin_amdgcn_s_wait_asynccnt(n)
#else
#define ASYNC_WAIT(n) asm volatile("s_wait_asynccnt %0" ::"i"(n) : "memory")
#endif
#else
#define USE_ASYNC_LDS 0
#define ASYNC_WAIT(n)
#endif

static __device__ __forceinline__ f32x8 wmma_bf16(bf16x16 a, bf16x16 b, f32x8 c) {
  return __builtin_amdgcn_wmma_f32_16x16x32_bf16(false, a, false, b, (short)0, c, false, false);
}

// A fragment, 16x32 bf16, from row-major tile (row m, 32 K cols).
// Lane layout (ISA 7.12.2): m = lane&15; lanes 0-15 K=0..7,16..23; lanes 16-31 K=8..15,24..31.
static __device__ __forceinline__ bf16x16 frag_a(const bf16* t, int ld, int lane) {
  const int m = lane & 15, h = (lane >> 4) << 3;
  bf16x16 a;
#pragma unroll
  for (int e = 0; e < 8; ++e) {
    a[e]     = t[m * ld + h + e];
    a[e + 8] = t[m * ld + 16 + h + e];
  }
  return a;
}

// B fragment, 32x16 bf16, from K-major tile (row = k, col = n). n = lane&15,
// K ordering mirrors the A fragment.
static __device__ __forceinline__ bf16x16 frag_b_k(const bf16* t, int ld, int lane) {
  const int n = lane & 15, h = (lane >> 4) << 3;
  bf16x16 b;
#pragma unroll
  for (int e = 0; e < 8; ++e) {
    b[e]     = t[(h + e) * ld + n];
    b[e + 8] = t[(16 + h + e) * ld + n];
  }
  return b;
}
// B from an N-major tile (row = n, col = k), i.e. B = Wᵀ with W row-major, uses
// frag_a's addressing (the lane's n indexes the row).

// ---------------------------------------------------------------------------
// Kernel 1: qkv = x @ qkv_w^T + qkv_b, scattered to bf16 Q/K/V [b*h][4096][64]
// ---------------------------------------------------------------------------
__global__ __launch_bounds__(128) void qkv_kernel(
    const float* __restrict__ x, const float* __restrict__ w,
    const float* __restrict__ bias,
    bf16* __restrict__ qw, bf16* __restrict__ kw, bf16* __restrict__ vw) {
  __shared__ bf16 As[64 * LDA];
  __shared__ bf16 Ws[64 * LDA];
  const int tid = threadIdx.x, lane = tid & 31, wv = tid >> 5;
  const int row = tid >> 1, halfc = (tid & 1) * 16;
  const int m0 = blockIdx.y * 64, n0 = blockIdx.x * 64;
  f32x8 acc[4] = {};

  for (int k0 = 0; k0 < 512; k0 += 32) {
    const float4* xs = (const float4*)(x + (size_t)(m0 + row) * 512 + k0 + halfc);
    const float4* ws = (const float4*)(w + (size_t)(n0 + row) * 512 + k0 + halfc);
#pragma unroll
    for (int i = 0; i < 4; ++i) {
      float4 xa = xs[i], wa = ws[i];
      bf16* da = &As[row * LDA + halfc + i * 4];
      bf16* dw = &Ws[row * LDA + halfc + i * 4];
      da[0] = (bf16)xa.x; da[1] = (bf16)xa.y; da[2] = (bf16)xa.z; da[3] = (bf16)xa.w;
      dw[0] = (bf16)wa.x; dw[1] = (bf16)wa.y; dw[2] = (bf16)wa.z; dw[3] = (bf16)wa.w;
    }
    __syncthreads();
    bf16x16 a = frag_a(&As[(wv * 16) * LDA], LDA, lane);
#pragma unroll
    for (int j = 0; j < 4; ++j) {
      bf16x16 b = frag_a(&Ws[(j * 16) * LDA], LDA, lane);  // Wᵀ: N-major == frag_a addressing
      acc[j] = wmma_bf16(a, b, acc[j]);
    }
    __syncthreads();
  }

  const int n = lane & 15, hh = lane >> 4;
#pragma unroll
  for (int j = 0; j < 4; ++j) {
    const int col = n0 + j * 16 + n;              // 0..1535 -> (head, which, cc)
    const float bs = bias[col];
    const int head = col / 192, rem = col % 192;
    const int which = rem >> 6, cc = rem & 63;
    bf16* dst = (which == 0) ? qw : ((which == 1) ? kw : vw);
#pragma unroll
    for (int v = 0; v < 8; ++v) {
      const int r = m0 + wv * 16 + v + hh * 8;    // global row in [0, 8192)
      const int bb = r >> 12, si = r & 4095;
      dst[((size_t)(bb * 8 + head) * 4096 + si) * 64 + cc] = (bf16)(acc[j][v] + bs);
    }
  }
}

// ---------------------------------------------------------------------------
// Kernel 2: flash attention. grid = (s/64, b*h), 4 waves x 16 query rows.
// K/V tiles double-buffered via async LDS loads (ASYNCcnt pipeline).
// ---------------------------------------------------------------------------
__global__ __launch_bounds__(128) void attn_kernel(
    const bf16* __restrict__ qw, const bf16* __restrict__ kw,
    const bf16* __restrict__ vw, bf16* __restrict__ ctx) {
  __shared__ bf16 kt[2][32 * LDK];
  __shared__ bf16 vt[2][32 * LDK];
  __shared__ bf16 ps[4][16 * LDP];
  const int tid = threadIdx.x, lane = tid & 31, wv = tid >> 5;
  const int bh = blockIdx.y;
  const int q0 = blockIdx.x * 64 + wv * 16;
  const bf16* kb = kw + (size_t)bh * 4096 * 64;
  const bf16* vb = vw + (size_t)bh * 4096 * 64;

  // Q fragments (16x64 -> two 16x32), scaled by 1/sqrt(64)=0.125 (exact in bf16)
  bf16x16 qf[2];
  {
    const bf16* qb = qw + ((size_t)bh * 4096 + q0) * 64;
    const int m = lane & 15, h = (lane >> 4) << 3;
#pragma unroll
    for (int c = 0; c < 2; ++c)
#pragma unroll
      for (int e = 0; e < 8; ++e) {
        qf[c][e]     = (bf16)((float)qb[m * 64 + c * 32 + h + e] * 0.125f);
        qf[c][e + 8] = (bf16)((float)qb[m * 64 + c * 32 + 16 + h + e] * 0.125f);
      }
  }

  f32x8 acc[4] = {};
  float mi[8], li[8];
#pragma unroll
  for (int v = 0; v < 8; ++v) { mi[v] = -1e30f; li[v] = 0.f; }
  const int hh = lane >> 4, n = lane & 15;

  // Stage one 32x64 K tile + V tile into buffer bufi (4 async ops / thread).
  auto stage = [&](int kv, int bufi) {
#if USE_ASYNC_LDS
    const int c0 = tid * 2;                      // 16B chunks; tile = 256 chunks
#pragma unroll
    for (int i = 0; i < 2; ++i) {
      const int c = c0 + i, r = c >> 3, col = (c & 7) * 8;
      ASYNC_LD16(kb + (size_t)(kv + r) * 64 + col, &kt[bufi][r * LDK + col]);
      ASYNC_LD16(vb + (size_t)(kv + r) * 64 + col, &vt[bufi][r * LDK + col]);
    }
#else
    const int r = tid >> 2, col = (tid & 3) * 16;
    const uint2* ks = (const uint2*)(kb + (size_t)(kv + r) * 64 + col);
    const uint2* vs = (const uint2*)(vb + (size_t)(kv + r) * 64 + col);
    uint2* kd = (uint2*)&kt[bufi][r * LDK + col];
    uint2* vd = (uint2*)&vt[bufi][r * LDK + col];
#pragma unroll
    for (int i = 0; i < 4; ++i) { kd[i] = ks[i]; vd[i] = vs[i]; }
#endif
  };

  stage(0, 0);
  for (int kv = 0; kv < 4096; kv += 32) {
    const int buf = (kv >> 5) & 1;
    if (kv + 32 < 4096) {
      stage(kv + 32, buf ^ 1);   // prefetch next tile into the other buffer
      ASYNC_WAIT(4);             // async loads retire in order: tile `kv` has landed
    } else {
      ASYNC_WAIT(0);
    }
    __syncthreads();

    // S = (Q/8) Kᵀ : K-dim is c=64 (two chunks), N = 32 keys (two groups)
    f32x8 s0 = {}, s1 = {};
#pragma unroll
    for (int c = 0; c < 2; ++c) {
      bf16x16 b0 = frag_a(&kt[buf][0 * LDK + c * 32], LDK, lane);   // keys 0..15
      bf16x16 b1 = frag_a(&kt[buf][16 * LDK + c * 32], LDK, lane);  // keys 16..31
      s0 = wmma_bf16(qf[c], b0, s0);
      s1 = wmma_bf16(qf[c], b1, s1);
    }

    // Online softmax: row m = v + 8*hh lives across the 16 lanes of a half-group
#pragma unroll
    for (int v = 0; v < 8; ++v) {
      float mx = fmaxf(s0[v], s1[v]);
#pragma unroll
      for (int d = 1; d < 16; d <<= 1) mx = fmaxf(mx, __shfl_xor(mx, d, 32));
      const float mnew  = fmaxf(mi[v], mx);
      const float alpha = __expf(mi[v] - mnew);
      const float p0 = __expf(s0[v] - mnew);
      const float p1 = __expf(s1[v] - mnew);
      float rs = p0 + p1;
#pragma unroll
      for (int d = 1; d < 16; d <<= 1) rs += __shfl_xor(rs, d, 32);
      li[v] = li[v] * alpha + rs;
      mi[v] = mnew;
      s0[v] = p0; s1[v] = p1;
#pragma unroll
      for (int j = 0; j < 4; ++j) acc[j][v] *= alpha;
    }

    // D-layout P -> wave-private LDS -> A-layout fragment
    {
      bf16* p = ps[wv];
#pragma unroll
      for (int v = 0; v < 8; ++v) {
        p[(v + hh * 8) * LDP + n]      = (bf16)s0[v];
        p[(v + hh * 8) * LDP + 16 + n] = (bf16)s1[v];
      }
    }
    asm volatile("s_wait_dscnt 0x0" ::: "memory");  // wave-local LDS RAW
    bf16x16 pf = frag_a(ps[wv], LDP, lane);
#pragma unroll
    for (int j = 0; j < 4; ++j) {
      bf16x16 vf = frag_b_k(&vt[buf][j * 16], LDK, lane);  // V: row=key (K-dim), col=c
      acc[j] = wmma_bf16(pf, vf, acc[j]);
    }
    __syncthreads();  // readers done before buffer `buf` is refilled at kv+64
  }

  // finalize: ctx[b*4096+q][head*64 + c] = acc / l
  const int b = bh >> 3, head = bh & 7;
#pragma unroll
  for (int v = 0; v < 8; ++v) {
    const float inv = 1.0f / li[v];
    const size_t r = (size_t)b * 4096 + q0 + v + hh * 8;
#pragma unroll
    for (int j = 0; j < 4; ++j)
      ctx[r * 512 + head * 64 + j * 16 + n] = (bf16)(acc[j][v] * inv);
  }
}

// ---------------------------------------------------------------------------
// Kernel 3: out = ctx @ out_w^T + out_b  (f32 output)
// ---------------------------------------------------------------------------
__global__ __launch_bounds__(128) void proj_kernel(
    const bf16* __restrict__ ctx, const float* __restrict__ w,
    const float* __restrict__ bias, float* __restrict__ out) {
  __shared__ bf16 As[64 * LDA];
  __shared__ bf16 Ws[64 * LDA];
  const int tid = threadIdx.x, lane = tid & 31, wv = tid >> 5;
  const int row = tid >> 1, halfc = (tid & 1) * 16;
  const int m0 = blockIdx.y * 64, n0 = blockIdx.x * 64;
  f32x8 acc[4] = {};

  for (int k0 = 0; k0 < 512; k0 += 32) {
    // A tile: bf16 ctx -> LDS. Async copy (no VGPR bounce) overlapped with the
    // synchronous f32->bf16 weight conversion below.
#if USE_ASYNC_LDS
    ASYNC_LD16(ctx + (size_t)(m0 + row) * 512 + k0 + halfc,     &As[row * LDA + halfc]);
    ASYNC_LD16(ctx + (size_t)(m0 + row) * 512 + k0 + halfc + 8, &As[row * LDA + halfc + 8]);
#else
    {
      const uint2* as = (const uint2*)(ctx + (size_t)(m0 + row) * 512 + k0 + halfc);
      uint2* ad = (uint2*)&As[row * LDA + halfc];
#pragma unroll
      for (int i = 0; i < 4; ++i) ad[i] = as[i];
    }
#endif
    {
      const float4* ws = (const float4*)(w + (size_t)(n0 + row) * 512 + k0 + halfc);
#pragma unroll
      for (int i = 0; i < 4; ++i) {
        float4 wa = ws[i];
        bf16* dw = &Ws[row * LDA + halfc + i * 4];
        dw[0] = (bf16)wa.x; dw[1] = (bf16)wa.y; dw[2] = (bf16)wa.z; dw[3] = (bf16)wa.w;
      }
    }
    ASYNC_WAIT(0);
    __syncthreads();
    bf16x16 a = frag_a(&As[(wv * 16) * LDA], LDA, lane);
#pragma unroll
    for (int j = 0; j < 4; ++j) {
      bf16x16 b = frag_a(&Ws[(j * 16) * LDA], LDA, lane);
      acc[j] = wmma_bf16(a, b, acc[j]);
    }
    __syncthreads();
  }

  const int n = lane & 15, hh = lane >> 4;
#pragma unroll
  for (int j = 0; j < 4; ++j) {
    const int col = n0 + j * 16 + n;
    const float bs = bias[col];
#pragma unroll
    for (int v = 0; v < 8; ++v) {
      const size_t r = (size_t)m0 + wv * 16 + v + hh * 8;
      out[r * 512 + col] = acc[j][v] + bs;
    }
  }
}

// ---------------------------------------------------------------------------
extern "C" void kernel_launch(void* const* d_in, const int* in_sizes, int n_in,
                              void* d_out, int out_size, void* d_ws, size_t ws_size,
                              hipStream_t stream) {
  const float* x     = (const float*)d_in[0];
  const float* qkv_w = (const float*)d_in[1];
  const float* qkv_b = (const float*)d_in[2];
  const float* out_w = (const float*)d_in[3];
  const float* out_b = (const float*)d_in[4];
  float* out = (float*)d_out;

  char* ws = (char*)d_ws;
  const size_t TSZ = (size_t)16 * 4096 * 64 * sizeof(bf16);  // 8 MB per Q/K/V tensor
  bf16* qw  = (bf16*)(ws);
  bf16* kw  = (bf16*)(ws + TSZ);
  bf16* vw  = (bf16*)(ws + 2 * TSZ);
  bf16* ctx = (bf16*)(ws + 3 * TSZ);

  qkv_kernel<<<dim3(24, 128), 128, 0, stream>>>(x, qkv_w, qkv_b, qw, kw, vw);
  attn_kernel<<<dim3(64, 16), 128, 0, stream>>>(qw, kw, vw, ctx);
  proj_kernel<<<dim3(8, 128), 128, 0, stream>>>(ctx, out_w, out_b, out);
}
